// GATNetwork_23038204576292
// MI455X (gfx1250) — matrix operand order repaired and checked by reading, not compiled, not run on hardware
//
#include <hip/hip_runtime.h>
#include <hip/hip_bf16.h>
#include <cstdint>

typedef __attribute__((ext_vector_type(16))) _Float16 v16h;
typedef __attribute__((ext_vector_type(8)))  float    v8f;

#define HC 128          // H*C = 4*32
#define NEG_SLOPE 0.2f

// ---------- monotonic float <-> u32 encoding (for atomic segment-max) ----------
__device__ __forceinline__ unsigned fenc(float f) {
  unsigned u = __float_as_uint(f);
  return (u & 0x80000000u) ? ~u : (u | 0x80000000u);
}
__device__ __forceinline__ float fdec(unsigned u) {
  u = (u & 0x80000000u) ? (u & 0x7fffffffu) : ~u;
  return __uint_as_float(u);
}

// ---------- WMMA fragment loaders (CDNA5 16x16x32 f16 layouts) ----------
// A: 16x32 f16, row m = lane&15; lanes0-15: K 0..7 (v0..3) & 16..23 (v4..7);
//    lanes16-31: K 8..15 & 24..31.  Source is f32 row-major, converted on load.
__device__ __forceinline__ v16h load_a_f32(const float* __restrict__ A, int row,
                                           int k0, int ld, int lane) {
  const int koff = (lane >> 4) * 8;
  const float* p = A + (long)row * ld + k0 + koff;
  v16h a;
#pragma unroll
  for (int j = 0; j < 8; ++j) {
    a[j]     = (_Float16)p[j];
    a[8 + j] = (_Float16)p[16 + j];
  }
  return a;
}

// B: 32x16 f16, col n = lane&15; lanes0-15 hold K 0..7 & 16..23, lanes16-31 hold K 8..15 & 24..31.
// WT is the transposed weight (row = output col n, 128 contiguous K) in f16,
// so each fragment is two contiguous 16-byte loads per lane.
__device__ __forceinline__ v16h load_b_tr(const _Float16* __restrict__ WT,
                                          int k0, int n0, int lane) {
  const _Float16* p = WT + (n0 + (lane & 15)) * HC + k0 + (lane >> 4) * 8;
  v16h b;
#pragma unroll
  for (int j = 0; j < 8; ++j) { b[j] = p[j]; b[8 + j] = p[16 + j]; }
  return b;
}

// ---------- 0) zero scratch ----------
__global__ void init_ws_kernel(unsigned* __restrict__ p, long n) {
  long i = (long)blockIdx.x * blockDim.x + threadIdx.x;
  const long stride = (long)gridDim.x * blockDim.x;
  for (; i < n; i += stride) p[i] = 0u;
}

// ---------- 1) transpose + f32->f16 convert the three 128x128 weights ----------
__global__ void prep_weights_kernel(const float* __restrict__ Wl,
                                    const float* __restrict__ Wr,
                                    const float* __restrict__ Wm,
                                    _Float16* __restrict__ WlT,
                                    _Float16* __restrict__ WrT,
                                    _Float16* __restrict__ WmT) {
  int idx = blockIdx.x * blockDim.x + threadIdx.x;
  if (idx < HC * HC) {
    int k = idx >> 7, n = idx & 127;
    int t = n * HC + k;
    WlT[t] = (_Float16)Wl[idx];
    WrT[t] = (_Float16)Wr[idx];
    WmT[t] = (_Float16)Wm[idx];
  }
}

// ---------- 2) fused xl = x@Wl, xr = x@Wr via WMMA ----------
// Block = 256 threads = 8 waves; block covers 128 rows x 128 cols.
// Wave w: rows m0 = blk*128 + w*16, loops all 8 col-tiles; A frags reused.
__global__ __launch_bounds__(256) void gatv2_xform_kernel(
    const float* __restrict__ x, const _Float16* __restrict__ WlT,
    const _Float16* __restrict__ WrT, float* __restrict__ xl,
    float* __restrict__ xr, int N) {
  const int lane = threadIdx.x & 31;
  const int wave = threadIdx.x >> 5;
  const int m0 = blockIdx.x * 128 + wave * 16;
  int arow = m0 + (lane & 15);
  if (arow >= N) arow = N - 1;  // clamp for A loads; stores are guarded

  v16h afrag[4];
#pragma unroll
  for (int kt = 0; kt < 4; ++kt) afrag[kt] = load_a_f32(x, arow, kt * 32, HC, lane);

#pragma unroll
  for (int nt = 0; nt < 8; ++nt) {
    v8f cl = {}; v8f cr = {};
#pragma unroll
    for (int kt = 0; kt < 4; ++kt) {
      v16h bl = load_b_tr(WlT, kt * 32, nt * 16, lane);
      v16h br = load_b_tr(WrT, kt * 32, nt * 16, lane);
      cl = __builtin_amdgcn_wmma_f32_16x16x32_f16(false, afrag[kt], false, bl,
                                                  (short)0, cl, false, false);
      cr = __builtin_amdgcn_wmma_f32_16x16x32_f16(false, afrag[kt], false, br,
                                                  (short)0, cr, false, false);
    }
    const int col = nt * 16 + (lane & 15);
    const int rbase = m0 + (lane >> 4) * 8;
#pragma unroll
    for (int i = 0; i < 8; ++i) {
      const int r = rbase + i;
      if (r < N) {
        xl[(long)r * HC + col] = cl[i];
        xr[(long)r * HC + col] = cr[i];
      }
    }
  }
}

// ---------- 3) per-edge logits + segment max (wave32 per edge) ----------
__global__ __launch_bounds__(256) void edge_logits_kernel(
    const int* __restrict__ srcv, const int* __restrict__ dstv,
    const float* __restrict__ xl, const float* __restrict__ xr,
    const float* __restrict__ att, float* __restrict__ logits,
    unsigned* __restrict__ m_enc, int E, int Etot) {
  const int e = blockIdx.x * 8 + (threadIdx.x >> 5);
  if (e >= Etot) return;
  const int lane = threadIdx.x & 31;
  int s, d;
  if (e < E) { s = srcv[e]; d = dstv[e]; } else { s = d = e - E; }
  const float* pl = xl + (long)s * HC;
  const float* pr = xr + (long)d * HC;
  float acc[4];
#pragma unroll
  for (int h = 0; h < 4; ++h) {
    float v = pl[h * 32 + lane] + pr[h * 32 + lane];
    v = (v > 0.f) ? v : NEG_SLOPE * v;       // leaky_relu (GATv2)
    acc[h] = v * att[h * 32 + lane];
  }
#pragma unroll
  for (int off = 16; off > 0; off >>= 1) {
#pragma unroll
    for (int h = 0; h < 4; ++h) acc[h] += __shfl_xor(acc[h], off, 32);
  }
  if (lane < 4) {
    logits[(long)e * 4 + lane] = acc[lane];
    atomicMax(&m_enc[d * 4 + lane], fenc(acc[lane]));
  }
}

// ---------- 4) ex = exp(logit - m[dst]); denom segment sum (thread per (e,h)) ----------
__global__ void edge_exp_kernel(const int* __restrict__ dstv,
                                float* __restrict__ logits,   // in: logits, out: ex
                                const unsigned* __restrict__ m_enc,
                                float* __restrict__ denom, int E, int Etot) {
  const long idx = (long)blockIdx.x * blockDim.x + threadIdx.x;
  if (idx >= (long)Etot * 4) return;
  const int e = (int)(idx >> 2), h = (int)(idx & 3);
  const int d = (e < E) ? dstv[e] : e - E;
  const float ex = __expf(logits[idx] - fdec(m_enc[d * 4 + h]));
  logits[idx] = ex;
  atomicAdd(&denom[d * 4 + h], ex);
}

// ---------- 5) out[dst] += alpha * xl[src] (wave32 per edge, L2 f32 atomics) ----------
__global__ __launch_bounds__(256) void edge_scatter_kernel(
    const int* __restrict__ srcv, const int* __restrict__ dstv,
    const float* __restrict__ ex, const float* __restrict__ denom,
    const float* __restrict__ xl, float* __restrict__ outacc, int E, int Etot) {
  const int e = blockIdx.x * 8 + (threadIdx.x >> 5);
  if (e >= Etot) return;
  const int lane = threadIdx.x & 31;
  int s, d;
  if (e < E) { s = srcv[e]; d = dstv[e]; } else { s = d = e - E; }
#pragma unroll
  for (int h = 0; h < 4; ++h) {
    const float alpha = ex[(long)e * 4 + h] / denom[d * 4 + h];
    const int ch = h * 32 + lane;
    atomicAdd(&outacc[(long)d * HC + ch], alpha * xl[(long)s * HC + ch]);
  }
}

// ---------- 6) +bias, relu, per-graph max pool (raw bits: values are >= 0) ----------
__global__ void bias_pool_kernel(const float* __restrict__ outacc,
                                 const float* __restrict__ bias,
                                 const int* __restrict__ batch_ids,
                                 unsigned* __restrict__ pooled_bits, int N) {
  const long idx = (long)blockIdx.x * blockDim.x + threadIdx.x;
  if (idx >= (long)N * HC) return;
  const int n = (int)(idx >> 7), ch = (int)(idx & 127);
  float v = outacc[idx] + bias[ch];
  v = v > 0.f ? v : 0.f;
  atomicMax(&pooled_bits[batch_ids[n] * HC + ch], __float_as_uint(v));
}

// ---------- 7) relu(pooled @ W_mlp + b) via WMMA, fused store to d_out ----------
__global__ __launch_bounds__(256) void mlp_kernel(
    const unsigned* __restrict__ pooled_bits, const _Float16* __restrict__ WmT,
    const float* __restrict__ bmlp, float* __restrict__ out, int G) {
  const int lane = threadIdx.x & 31;
  const int wave = threadIdx.x >> 5;
  const int m0 = blockIdx.x * 64 + (wave >> 1) * 16;
  const int ntbase = (wave & 1) * 4;

  int row = m0 + (lane & 15);
  if (row >= G) row = G - 1;
  const unsigned* p = pooled_bits + row * HC;
  const int koff = (lane >> 4) * 8;
  v16h afrag[4];
#pragma unroll
  for (int kt = 0; kt < 4; ++kt) {
#pragma unroll
    for (int j = 0; j < 8; ++j) {
      afrag[kt][j]     = (_Float16)__uint_as_float(p[kt * 32 + koff + j]);
      afrag[kt][8 + j] = (_Float16)__uint_as_float(p[kt * 32 + 16 + koff + j]);
    }
  }
#pragma unroll
  for (int t = 0; t < 4; ++t) {
    const int n0 = (ntbase + t) * 16;
    v8f c = {};
#pragma unroll
    for (int kt = 0; kt < 4; ++kt) {
      v16h b = load_b_tr(WmT, kt * 32, n0, lane);
      c = __builtin_amdgcn_wmma_f32_16x16x32_f16(false, afrag[kt], false, b,
                                                 (short)0, c, false, false);
    }
    const int col = n0 + (lane & 15);
    const int rbase = m0 + (lane >> 4) * 8;
#pragma unroll
    for (int i = 0; i < 8; ++i) {
      const int r = rbase + i;
      if (r < G) {
        const float v = c[i] + bmlp[col];
        out[r * HC + col] = v > 0.f ? v : 0.f;
      }
    }
  }
}

extern "C" void kernel_launch(void* const* d_in, const int* in_sizes, int n_in,
                              void* d_out, int out_size, void* d_ws, size_t ws_size,
                              hipStream_t stream) {
  (void)n_in; (void)ws_size;
  const float* x         = (const float*)d_in[0];
  const int*   edge_idx  = (const int*)d_in[1];   // [2,E] flat: src then dst
  const int*   batch_ids = (const int*)d_in[2];
  const float* Wl        = (const float*)d_in[4];
  const float* Wr        = (const float*)d_in[5];
  const float* att       = (const float*)d_in[6];
  const float* bias_conv = (const float*)d_in[7];
  const float* Wmlp      = (const float*)d_in[8];
  const float* bmlp      = (const float*)d_in[9];

  const int N    = in_sizes[0] / HC;
  const int E    = in_sizes[1] / 2;
  const int Etot = E + N;
  const int G    = out_size / HC;
  const int* srcv = edge_idx;
  const int* dstv = edge_idx + E;

  // ---- workspace layout (256B-aligned slices) ----
  char* ws = (char*)d_ws;
  auto take = [&](size_t bytes) {
    char* p = ws;
    ws += (bytes + 255) & ~(size_t)255;
    return p;
  };
  float*     xl     = (float*)take((size_t)N * HC * 4);
  float*     xr     = (float*)take((size_t)N * HC * 4);
  float*     logits = (float*)take((size_t)Etot * 4 * 4);   // reused as ex
  _Float16*  WlT    = (_Float16*)take(HC * HC * 2);
  _Float16*  WrT    = (_Float16*)take(HC * HC * 2);
  _Float16*  WmT    = (_Float16*)take(HC * HC * 2);
  // zero-initialized region (contiguous so one memset kernel covers it)
  char* zbeg = ws;
  unsigned* m_enc       = (unsigned*)take((size_t)N * 4 * 4);
  float*    denom       = (float*)take((size_t)N * 4 * 4);
  float*    outacc      = (float*)take((size_t)N * HC * 4);
  unsigned* pooled_bits = (unsigned*)take((size_t)G * HC * 4);
  const long zwords = (long)(ws - zbeg) / 4;

  init_ws_kernel<<<2048, 256, 0, stream>>>((unsigned*)zbeg, zwords);
  prep_weights_kernel<<<(HC * HC + 255) / 256, 256, 0, stream>>>(Wl, Wr, Wmlp,
                                                                 WlT, WrT, WmT);
  gatv2_xform_kernel<<<(N + 127) / 128, 256, 0, stream>>>(x, WlT, WrT, xl, xr, N);
  edge_logits_kernel<<<(Etot + 7) / 8, 256, 0, stream>>>(srcv, dstv, xl, xr, att,
                                                         logits, m_enc, E, Etot);
  edge_exp_kernel<<<(int)(((long)Etot * 4 + 255) / 256), 256, 0, stream>>>(
      dstv, logits, m_enc, denom, E, Etot);
  edge_scatter_kernel<<<(Etot + 7) / 8, 256, 0, stream>>>(srcv, dstv, logits, denom,
                                                          xl, outacc, E, Etot);
  bias_pool_kernel<<<(int)(((long)N * HC + 255) / 256), 256, 0, stream>>>(
      outacc, bias_conv, batch_ids, pooled_bits, N);
  mlp_kernel<<<(G + 63) / 64, 256, 0, stream>>>(pooled_bits, WmT, bmlp,
                                                (float*)d_out, G);
}
